// LSHTable_14216341749766
// MI455X (gfx1250) — 8-bit matrix operand order repaired and checked by reading, compiled, not run on hardware
//
#include <hip/hip_runtime.h>
#include <stdint.h>

#define B_    4
#define NQ_   2048
#define NP__  8192
#define D_    256
#define F_    40
#define KPAD  64
#define KOUT  16
#define QT    (NQ_ / 16)
#define PT    (NP__ / 16)
#define BINS  41
#define BINP  48

typedef int v8i __attribute__((ext_vector_type(8)));

__global__ __launch_bounds__(320) void lsh_hash_kernel(
    const float* __restrict__ x,
    const float* __restrict__ proj,
    uint32_t* __restrict__ tiles,
    uint2* __restrict__ bits,
    int blayout)
{
  __shared__ float s_x[8][D_ + 4];
  __shared__ unsigned char s_bytes[8][KPAD];

  const int tx  = threadIdx.x;
  const int il  = threadIdx.y;
  const int tid = il * 40 + tx;
  const long item0 = (long)blockIdx.x * 8;

  for (int idx = tid; idx < 8 * D_; idx += 320) {
    int r = idx >> 8, d = idx & (D_ - 1);
    s_x[r][d] = x[(item0 + r) * D_ + d];
  }
  __syncthreads();

  const float4* pr4 = (const float4*)(proj + (long)tx * D_);
  float acc = 0.f;
#pragma unroll 8
  for (int i = 0; i < D_ / 4; ++i) {
    float4 p = pr4[i];
    acc += s_x[il][4 * i + 0] * p.x;
    acc += s_x[il][4 * i + 1] * p.y;
    acc += s_x[il][4 * i + 2] * p.z;
    acc += s_x[il][4 * i + 3] * p.w;
  }
  s_bytes[il][tx] = (acc > 0.f) ? (unsigned char)1 : (unsigned char)0xFF;
  if (tx < KPAD - F_) s_bytes[il][F_ + tx] = 0;
  __syncthreads();

  if (tid < 128) {
    int j  = tid & 15;
    int r  = tid >> 4;
    int hf = j >> 3;
    int v  = j & 7;
    int k  = blayout ? (((v >> 1) << 4) + ((v & 1) << 2) + (hf << 3))
                     : (((v >> 1) << 4) + ((v & 1) << 2) + (hf << 3));
    uint32_t dw = (uint32_t)s_bytes[r][k]
                | ((uint32_t)s_bytes[r][k + 1] << 8)
                | ((uint32_t)s_bytes[r][k + 2] << 16)
                | ((uint32_t)s_bytes[r][k + 3] << 24);
    long item = item0 + r;
    long tile = item >> 4;
    int  row  = (int)(item & 15);
    int  lane = (hf << 4) | row;
    tiles[tile * 256 + lane * 8 + v] = dw;
  }

  if (tx == 0) {
    uint32_t w0 = 0, w1 = 0;
    for (int f = 0; f < 32; ++f)  if (s_bytes[il][f] == 1) w0 |= (1u << f);
    for (int f = 32; f < F_; ++f) if (s_bytes[il][f] == 1) w1 |= (1u << (f - 32));
    bits[item0 + il] = make_uint2(w0, w1);
  }
}

__device__ __forceinline__ v8i lsh_ld_tile(const uint4* t, long idx) {
  uint4 lo = t[idx + 0];
  uint4 hi = t[idx + 1];
  v8i r = {(int)lo.x, (int)lo.y, (int)lo.z, (int)lo.w,
           (int)hi.x, (int)hi.y, (int)hi.z, (int)hi.w};
  return r;
}

__device__ __forceinline__ void lsh_hist8(char* hrow, v8i acc) {
#pragma unroll
  for (int r = 0; r < 8; ++r) {
    int dist = (F_ - acc[r]) >> 1;
    dist = dist < 0 ? 0 : (dist > BINS - 1 ? BINS - 1 : dist);
    atomicAdd((unsigned int*)(hrow + (size_t)r * (BINP * 4)) + dist, 1u);
  }
}

__global__ __launch_bounds__(256) void lsh_topk_kernel(
    const uint32_t* __restrict__ aq_tiles,
    const uint32_t* __restrict__ bp_tiles,
    const uint2* __restrict__ qbits,
    const uint2* __restrict__ pbits,
    float* __restrict__ out_idx,
    float* __restrict__ out_val)
{
  __shared__ unsigned int hist[16][BINP];
  __shared__ unsigned int cnt[16][BINP];
  __shared__ unsigned int thr[16];

  const int tid  = threadIdx.x;
  const int lane = tid & 31;
  const int wave = tid >> 5;
  const int wg   = blockIdx.x;
  const int b    = wg / QT;

  for (int i = tid; i < 16 * BINP; i += 256) {
    (&hist[0][0])[i] = 0;
    (&cnt[0][0])[i]  = 0;
  }
  __syncthreads();

  const uint4* aq4 = (const uint4*)aq_tiles;
  const uint4* bp4 = (const uint4*)bp_tiles;
  uint4 alo = aq4[(long)wg * 64 + lane * 2 + 0];
  uint4 ahi = aq4[(long)wg * 64 + lane * 2 + 1];
  v8i A = {(int)alo.x, (int)alo.y, (int)alo.z, (int)alo.w,
           (int)ahi.x, (int)ahi.y, (int)ahi.z, (int)ahi.w};
  const v8i Cz = {0, 0, 0, 0, 0, 0, 0, 0};
  const int mrow = (lane >> 4) * 8;
  char* const hrow = (char*)&hist[mrow][0];

  for (int pt = wave; pt < PT; pt += 32) {
    long b0 = ((long)(b * PT + pt)) * 64 + lane * 2;
    v8i B0 = lsh_ld_tile(bp4, b0);
    v8i B1 = lsh_ld_tile(bp4, b0 + 8 * 64);
    v8i B2 = lsh_ld_tile(bp4, b0 + 16 * 64);
    v8i B3 = lsh_ld_tile(bp4, b0 + 24 * 64);
    v8i a0 = __builtin_amdgcn_wmma_i32_16x16x64_iu8(true, A, true, B0, Cz,
                                                    false, false);
    v8i a1 = __builtin_amdgcn_wmma_i32_16x16x64_iu8(true, A, true, B1, Cz,
                                                    false, false);
    v8i a2 = __builtin_amdgcn_wmma_i32_16x16x64_iu8(true, A, true, B2, Cz,
                                                    false, false);
    v8i a3 = __builtin_amdgcn_wmma_i32_16x16x64_iu8(true, A, true, B3, Cz,
                                                    false, false);
    lsh_hist8(hrow, a0);
    lsh_hist8(hrow, a1);
    lsh_hist8(hrow, a2);
    lsh_hist8(hrow, a3);
  }
  __syncthreads();

  if (tid < 16) {
    unsigned run = 0, T = BINS;
    for (int d = 0; d < BINS; ++d) {
      unsigned c = hist[tid][d];
      hist[tid][d] = run;
      run += c;
      if (T == BINS && run >= KOUT) T = (unsigned)d;
    }
    thr[tid] = T;
  }
  __syncthreads();

  for (int qi = 0; qi < 2; ++qi) {
    int  q  = wave * 2 + qi;
    long qg = (long)wg * 16 + q;
    uint2 qb = qbits[qg];
    int  T  = (int)thr[q];
    for (int p0 = 0; p0 < NP__; p0 += 32) {
      int   p  = p0 + lane;
      uint2 pb = pbits[(long)b * NP__ + p];
      int d = __popc(qb.x ^ pb.x) + __popc(qb.y ^ pb.y);
      unsigned long long m = __ballot(d <= T);
      while (m) {
        int l = __builtin_ctzll(m);
        if (lane == l) {
          unsigned slot = hist[q][d] + atomicAdd(&cnt[q][d], 1u);
          if (slot < KOUT) {
            long o = qg * KOUT + slot;
            out_idx[o] = (float)p;
            out_val[o] = (float)d;
          }
        }
        m &= (m - 1);
      }
    }
  }
}

extern "C" void kernel_launch(void* const* d_in, const int* in_sizes, int n_in,
                              void* d_out, int out_size, void* d_ws,
                              size_t ws_size, hipStream_t stream) {
  (void)in_sizes; (void)n_in; (void)out_size; (void)ws_size;
  const float* qpts = (const float*)d_in[0];
  const float* pts  = (const float*)d_in[1];
  const float* proj = (const float*)d_in[2];

  char* ws = (char*)d_ws;
  uint32_t* aq_tiles = (uint32_t*)ws;
  uint32_t* bp_tiles = (uint32_t*)(ws + (size_t)512 * 1024);
  uint2*    qbits    = (uint2*)(ws + (size_t)(512 + 2048) * 1024);
  uint2*    pbits    = (uint2*)((char*)qbits + (size_t)B_ * NQ_ * 8);

  lsh_hash_kernel<<<dim3(B_ * NQ_ / 8), dim3(40, 8), 0, stream>>>(
      qpts, proj, aq_tiles, qbits,  0);
  lsh_hash_kernel<<<dim3(B_ * NP__ / 8), dim3(40, 8), 0, stream>>>(
      pts, proj, bp_tiles, pbits,  1);

  float* out_idx = (float*)d_out;
  float* out_val = out_idx + (size_t)B_ * NQ_ * KOUT;
  lsh_topk_kernel<<<dim3(B_ * QT), dim3(256), 0, stream>>>(
      aq_tiles, bp_tiles, qbits, pbits, out_idx, out_val);
}
